// ResMoELoRALinear_1864015807037
// MI455X (gfx1250) — compile-verified, matrix operand order checked by reading, not run on hardware
//
#include <hip/hip_runtime.h>
#include <cstdint>
#include <cstddef>

#define IN_F  2048
#define OUT_F 2048
#define RES   64
#define NEXP  16
#define NTOK  16384   // 4 * 4096

typedef __bf16 bf16;
typedef bf16  v16bf __attribute__((ext_vector_type(16)));
typedef bf16  v8bf  __attribute__((ext_vector_type(8)));
typedef float v8f   __attribute__((ext_vector_type(8)));

#define PITCH  72    // LDS row pitch in halves (64 data + 8 pad)
#define PITCHB 144   // bytes

#define WMMA_BF16(a, b, c) \
  __builtin_amdgcn_wmma_f32_16x16x32_bf16(false, (a), false, (b), (short)0, (c), false, false)

// sched_group_barrier masks
#define SGB_WMMA    0x008
#define SGB_DS_READ 0x100

// ---------------------------------------------------------------------------
// CDNA5 async-copy helpers (global -> LDS, tracked by ASYNCcnt)
// ---------------------------------------------------------------------------
__device__ __forceinline__ unsigned lds_off(const void* p) {
  // generic shared pointer: low 32 bits == LDS byte address
  return (unsigned)(uintptr_t)p;
}

__device__ __forceinline__ void async_b128(unsigned lds, const void* g) {
  asm volatile("global_load_async_to_lds_b128 %0, %1, off" :: "v"(lds), "v"(g) : "memory");
}

__device__ __forceinline__ void wait_async0() {
  asm volatile("s_wait_asynccnt 0" ::: "memory");
}

// stage ROWS rows of 64 bf16 (128B each) from gsrc (row stride in halves)
// into LDS at byte pitch PITCHB; 256 threads, 16B async chunk per op.
template <int ROWS>
__device__ __forceinline__ void stage_rows(unsigned lds_base, const bf16* gsrc,
                                           int stride_halves, int tid) {
  constexpr int TOTAL = ROWS * 8;               // 16B chunks
  constexpr int ITERS = (TOTAL + 255) / 256;
#pragma unroll
  for (int i = 0; i < ITERS; ++i) {
    const int idx = tid + i * 256;
    if ((TOTAL % 256 == 0) || (idx < TOTAL)) {
      const int r = idx >> 3;
      const int c = (idx & 7) << 4;             // byte offset within row
      async_b128(lds_base + (unsigned)(r * PITCHB + c),
                 (const char*)(gsrc + (size_t)r * stride_halves) + c);
    }
  }
}

template <int NBYTES>
__device__ __forceinline__ void stage_linear(unsigned lds_base, const void* gsrc, int tid) {
  constexpr int TOTAL = NBYTES >> 4;
  constexpr int ITERS = (TOTAL + 255) / 256;
#pragma unroll
  for (int i = 0; i < ITERS; ++i) {
    const int idx = tid + i * 256;
    if ((TOTAL % 256 == 0) || (idx < TOTAL))
      async_b128(lds_base + (unsigned)(idx << 4), (const char*)gsrc + (idx << 4));
  }
}

// ---------------------------------------------------------------------------
// WMMA fragment load from pitched LDS tile (16-bit A/B layout, ISA 7.12.2):
// lane L: row = L&15; halves 0..7  -> K = kk + (L>=16 ? 8 : 0) .. +7
//                     halves 8..15 -> K = kk + (L>=16 ? 8 : 0) + 16 .. +23
// ---------------------------------------------------------------------------
__device__ __forceinline__ v16bf frag_ld(const bf16* rowp, int kk, int hi8) {
  v8bf lo = *(const v8bf*)(rowp + kk + hi8);
  v8bf hi = *(const v8bf*)(rowp + kk + hi8 + 16);
  return __builtin_shufflevector(lo, hi, 0, 1, 2, 3, 4, 5, 6, 7,
                                         8, 9, 10, 11, 12, 13, 14, 15);
}

// row-scale an A fragment (one M-row per lane) -> v_pk_mul_bf16
__device__ __forceinline__ v16bf frag_scale(v16bf f, float s) {
  return f * (bf16)s;   // vector * scalar splat
}

// ---------------------------------------------------------------------------
// Pass 0: one-time f32 -> bf16 down-conversions
// ---------------------------------------------------------------------------
__global__ void cvt_f32_bf16(const float* __restrict__ src, bf16* __restrict__ dst, int n) {
  int i = blockIdx.x * 256 + threadIdx.x;
  if (i < n) dst[i] = (bf16)src[i];
}

// concat [A (64 x 2048) ; router_W (16 x 2048)] -> Pb [80 x 2048] bf16
__global__ void build_P(const float* __restrict__ A, const float* __restrict__ rw,
                        bf16* __restrict__ Pb) {
  int i = blockIdx.x * 256 + threadIdx.x;
  if (i < 80 * IN_F) {
    int r = i / IN_F, k = i - r * IN_F;
    float v = (r < RES) ? A[(size_t)r * IN_F + k] : rw[(size_t)(r - RES) * IN_F + k];
    Pb[i] = (bf16)v;
  }
}

// ---------------------------------------------------------------------------
// Pass 1: hl = x @ [A;router_W]^T  -> hb [N,64] bf16, logits [N,16] f32
// 128 tokens/WG, 8 waves x 16 tokens x 80 outs (5 wmma tiles), K=2048,
// double-buffered async pipeline.
// ---------------------------------------------------------------------------
__global__ __launch_bounds__(256) void reservoir_router_gemm(
    const bf16* __restrict__ xb, const bf16* __restrict__ Pb,
    bf16* __restrict__ hb, float* __restrict__ lg) {
  __shared__ bf16 Xs[2][128 * PITCH];
  __shared__ bf16 Ps[2][80 * PITCH];
  const int tid = threadIdx.x;
  const int lane = tid & 31, w = tid >> 5;
  const int r = lane & 15, hi8 = (lane >> 4) << 3;
  const int tileM = blockIdx.x * 128;
  const unsigned xs[2] = {lds_off(Xs[0]), lds_off(Xs[1])};
  const unsigned ps[2] = {lds_off(Ps[0]), lds_off(Ps[1])};

  v8f acc[5] = {};
  // prologue: stage tile 0
  stage_rows<128>(xs[0], xb + (size_t)tileM * IN_F, IN_F, tid);
  stage_rows<80>(ps[0], Pb, IN_F, tid);
  wait_async0();
  __syncthreads();

  int p = 0;
  for (int k0 = 0; k0 < IN_F; k0 += 64) {
    if (k0 + 64 < IN_F) {  // prefetch next K tile into other buffer
      stage_rows<128>(xs[1 - p], xb + (size_t)tileM * IN_F + k0 + 64, IN_F, tid);
      stage_rows<80>(ps[1 - p], Pb + k0 + 64, IN_F, tid);
    }
    // load all fragments of the K-tile first, then all WMMAs
    v16bf a[2], b[2][5];
#pragma unroll
    for (int kki = 0; kki < 2; ++kki) {
      a[kki] = frag_ld(&Xs[p][(w * 16 + r) * PITCH], kki * 32, hi8);
#pragma unroll
      for (int nt = 0; nt < 5; ++nt)
        b[kki][nt] = frag_ld(&Ps[p][(nt * 16 + r) * PITCH], kki * 32, hi8);
    }
#pragma unroll
    for (int kki = 0; kki < 2; ++kki)
#pragma unroll
      for (int nt = 0; nt < 5; ++nt)
        acc[nt] = WMMA_BF16(a[kki], b[kki][nt], acc[nt]);
    __builtin_amdgcn_sched_group_barrier(SGB_DS_READ, 24, 0);
    __builtin_amdgcn_sched_group_barrier(SGB_WMMA, 10, 0);
    wait_async0();
    __syncthreads();
    p ^= 1;
  }
  // C/D layout: lanes 0-15: M=v, N=lane; lanes 16-31: M=v+8, N=lane-16
#pragma unroll
  for (int v = 0; v < 8; ++v) {
    const int m = tileM + w * 16 + ((lane < 16) ? v : v + 8);
#pragma unroll
    for (int nt = 0; nt < 4; ++nt)
      hb[(size_t)m * RES + nt * 16 + r] = (bf16)acc[nt][v];
    lg[(size_t)m * NEXP + r] = acc[4][v];
  }
}

// ---------------------------------------------------------------------------
// Pass 2: softmax + top-2 + renorm -> dense wf [N,16]
// ---------------------------------------------------------------------------
__global__ void softmax_top2(const float* __restrict__ lg, float* __restrict__ wf) {
  int t = blockIdx.x * 256 + threadIdx.x;
  if (t >= NTOK) return;
  float l[NEXP];
  float mx = -3.4e38f;
#pragma unroll
  for (int i = 0; i < NEXP; ++i) { l[i] = lg[(size_t)t * NEXP + i]; mx = fmaxf(mx, l[i]); }
  float Z = 0.f;
#pragma unroll
  for (int i = 0; i < NEXP; ++i) { l[i] = __expf(l[i] - mx); Z += l[i]; }
  int i1 = 0; float p1 = -1.f;
#pragma unroll
  for (int i = 0; i < NEXP; ++i) if (l[i] > p1) { p1 = l[i]; i1 = i; }
  int i2 = 0; float p2 = -1.f;
#pragma unroll
  for (int i = 0; i < NEXP; ++i) if (i != i1 && l[i] > p2) { p2 = l[i]; i2 = i; }
  p1 /= Z; p2 /= Z;
  const float s = 1.f / (p1 + p2 + 1e-6f);
#pragma unroll
  for (int i = 0; i < NEXP; ++i) wf[(size_t)t * NEXP + i] = 0.f;
  wf[(size_t)t * NEXP + i1] = p1 * s;
  wf[(size_t)t * NEXP + i2] += p2 * s;
}

// ---------------------------------------------------------------------------
// Pass 3: out = x@W^T + b + sum_e (wf[:,e] * h) @ B[e]^T
// 128x128 tile/WG; 8 waves (4 M x 2 N), wave tile 32x64; double-buffered
// async pipeline for both the base K-loop and the 16-expert loop.
// ---------------------------------------------------------------------------
__device__ __forceinline__ void mma_tile(const bf16* Xp, const bf16* Wp,
                                         int wm, int wn, int r, int hi8,
                                         v8f (&acc)[2][4]) {
  v16bf a[2][2], b[2][4];
#pragma unroll
  for (int kki = 0; kki < 2; ++kki) {
    a[kki][0] = frag_ld(&Xp[(wm + r) * PITCH], kki * 32, hi8);
    a[kki][1] = frag_ld(&Xp[(wm + 16 + r) * PITCH], kki * 32, hi8);
#pragma unroll
    for (int nt = 0; nt < 4; ++nt)
      b[kki][nt] = frag_ld(&Wp[(wn + nt * 16 + r) * PITCH], kki * 32, hi8);
  }
#pragma unroll
  for (int kki = 0; kki < 2; ++kki)
#pragma unroll
    for (int nt = 0; nt < 4; ++nt) {
      acc[0][nt] = WMMA_BF16(a[kki][0], b[kki][nt], acc[0][nt]);
      acc[1][nt] = WMMA_BF16(a[kki][1], b[kki][nt], acc[1][nt]);
    }
  // all 24 DS reads of the K-tile first, then the 16 WMMAs:
  // first WMMA waits on a partial DScnt while the tail loads overlap the
  // matrix pipe instead of s_wait_dscnt 0 per WMMA.
  __builtin_amdgcn_sched_group_barrier(SGB_DS_READ, 24, 0);
  __builtin_amdgcn_sched_group_barrier(SGB_WMMA, 16, 0);
}

__global__ __launch_bounds__(256) void fused_moe_gemm(
    const bf16* __restrict__ xb, const bf16* __restrict__ wb,
    const bf16* __restrict__ Bb, const bf16* __restrict__ hb,
    const float* __restrict__ wf, const float* __restrict__ bias,
    float* __restrict__ out) {
  __shared__ bf16 Xs[2][128 * PITCH];
  __shared__ bf16 Ws[2][128 * PITCH];
  __shared__ float Wf[128 * NEXP];
  const int tid = threadIdx.x;
  const int lane = tid & 31, w = tid >> 5;
  const int r = lane & 15, hi8 = (lane >> 4) << 3;
  const int wm = (w & 3) * 32, wn = (w >> 2) * 64;
  const int tileM = blockIdx.x * 128, tileN = blockIdx.y * 128;
  const unsigned xs[2] = {lds_off(Xs[0]), lds_off(Xs[1])};
  const unsigned ws[2] = {lds_off(Ws[0]), lds_off(Ws[1])};
  const unsigned wfs = lds_off(Wf);

  v8f acc[2][4] = {};

  // ---- base GEMM pipeline ----
  stage_rows<128>(xs[0], xb + (size_t)tileM * IN_F, IN_F, tid);
  stage_rows<128>(ws[0], wb + (size_t)tileN * IN_F, IN_F, tid);
  wait_async0();
  __syncthreads();

  int p = 0;
  for (int k0 = 0; k0 < IN_F; k0 += 64) {
    if (k0 + 64 < IN_F) {  // prefetch next K tile
      stage_rows<128>(xs[1 - p], xb + (size_t)tileM * IN_F + k0 + 64, IN_F, tid);
      stage_rows<128>(ws[1 - p], wb + (size_t)tileN * IN_F + k0 + 64, IN_F, tid);
    } else {               // prefetch expert-phase h tile + router weights
      stage_rows<128>(xs[1 - p], hb + (size_t)tileM * RES, RES, tid);
      stage_linear<128 * NEXP * 4>(wfs, wf + (size_t)tileM * NEXP, tid);
    }
    mma_tile(Xs[p], Ws[p], wm, wn, r, hi8, acc);
    wait_async0();
    __syncthreads();
    p ^= 1;
  }

  // ---- expert phase: acc += (wf[:,e] * h) @ B[e]^T, K = 64 per expert ----
  // Xs[p] now holds the h tile; Wf holds the dense top-2 weights.
  v16bf hf[2][2];
#pragma unroll
  for (int mt = 0; mt < 2; ++mt)
#pragma unroll
    for (int kki = 0; kki < 2; ++kki)
      hf[mt][kki] = frag_ld(&Xs[p][(wm + mt * 16 + r) * PITCH], kki * 32, hi8);

  int q = 0;
  stage_rows<128>(ws[0], Bb + (size_t)tileN * RES, RES, tid);  // expert 0
  wait_async0();
  __syncthreads();

  for (int e = 0; e < NEXP; ++e) {
    if (e + 1 < NEXP)  // prefetch next expert's B tile
      stage_rows<128>(ws[1 - q], Bb + ((size_t)(e + 1) * OUT_F + tileN) * RES, RES, tid);
    const float s0 = Wf[(wm + r) * NEXP + e];
    const float s1 = Wf[(wm + 16 + r) * NEXP + e];
    v16bf b[2][4];
#pragma unroll
    for (int kki = 0; kki < 2; ++kki)
#pragma unroll
      for (int nt = 0; nt < 4; ++nt)
        b[kki][nt] = frag_ld(&Ws[q][(wn + nt * 16 + r) * PITCH], kki * 32, hi8);
#pragma unroll
    for (int kki = 0; kki < 2; ++kki) {
      v16bf a0 = frag_scale(hf[0][kki], s0);
      v16bf a1 = frag_scale(hf[1][kki], s1);
#pragma unroll
      for (int nt = 0; nt < 4; ++nt) {
        acc[0][nt] = WMMA_BF16(a0, b[kki][nt], acc[0][nt]);
        acc[1][nt] = WMMA_BF16(a1, b[kki][nt], acc[1][nt]);
      }
    }
    // 16 fragment reads + 2 scale reads pinned ahead of the 16 WMMAs;
    // v_pk_mul_bf16 scaling left free for the scheduler to interleave.
    __builtin_amdgcn_sched_group_barrier(SGB_DS_READ, 18, 0);
    __builtin_amdgcn_sched_group_barrier(SGB_WMMA, 16, 0);
    wait_async0();
    __syncthreads();
    q ^= 1;
  }

  // ---- epilogue: + bias, store f32 (lanes map to adjacent cols: coalesced) ----
#pragma unroll
  for (int nt = 0; nt < 4; ++nt) {
    const int col = tileN + wn + nt * 16 + r;
    const float bv = bias[col];
#pragma unroll
    for (int mt = 0; mt < 2; ++mt) {
#pragma unroll
      for (int v = 0; v < 8; ++v) {
        const int row = tileM + wm + mt * 16 + ((lane < 16) ? v : v + 8);
        out[(size_t)row * OUT_F + col] = acc[mt][nt][v] + bv;
      }
    }
  }
}

// ---------------------------------------------------------------------------
extern "C" void kernel_launch(void* const* d_in, const int* in_sizes, int n_in,
                              void* d_out, int out_size, void* d_ws, size_t ws_size,
                              hipStream_t stream) {
  (void)in_sizes; (void)n_in; (void)out_size; (void)ws_size;
  const float* x  = (const float*)d_in[0];
  const float* bw = (const float*)d_in[1];
  const float* bb = (const float*)d_in[2];
  const float* A  = (const float*)d_in[3];
  const float* B  = (const float*)d_in[4];
  const float* rw = (const float*)d_in[5];
  float* out = (float*)d_out;

  char* ws = (char*)d_ws;
  bf16*  xb = (bf16*)(ws);                      // 64 MB  x in bf16
  bf16*  wb = (bf16*)(ws + 67108864);           //  8 MB  base_W bf16
  bf16*  Bb = (bf16*)(ws + 75497472);           //  4 MB  B bf16
  bf16*  Pb = (bf16*)(ws + 79691776);           // 320 KB [A;router_W] bf16
  bf16*  hb = (bf16*)(ws + 80019456);           //  2 MB  h bf16
  float* lg = (float*)(ws + 82116608);          //  1 MB  logits
  float* wf = (float*)(ws + 83165184);          //  1 MB  dense top-2 weights

  cvt_f32_bf16<<<(NTOK * IN_F + 255) / 256, 256, 0, stream>>>(x, xb, NTOK * IN_F);
  cvt_f32_bf16<<<(OUT_F * IN_F + 255) / 256, 256, 0, stream>>>(bw, wb, OUT_F * IN_F);
  cvt_f32_bf16<<<(NEXP * OUT_F * RES + 255) / 256, 256, 0, stream>>>(B, Bb, NEXP * OUT_F * RES);
  build_P<<<(80 * IN_F + 255) / 256, 256, 0, stream>>>(A, rw, Pb);

  reservoir_router_gemm<<<NTOK / 128, 256, 0, stream>>>(xb, Pb, hb, lg);
  softmax_top2<<<NTOK / 256, 256, 0, stream>>>(lg, wf);

  dim3 grid(NTOK / 128, OUT_F / 128);
  fused_moe_gemm<<<grid, 256, 0, stream>>>(xb, wb, Bb, hb, wf, bb, out);
}
